// ParameterizedExperts_18176301597237
// MI455X (gfx1250) — matrix operand order repaired and checked by reading, not compiled, not run on hardware
//
#include <hip/hip_runtime.h>

// Grouped (expert) GEMM: out[e] = X[e] (2048x512) @ W[e]^T (512x512), fp32.
// bf16x3 split-precision WMMA (Ahi*Bhi + Ahi*Blo + Alo*Bhi), fp32 accumulate,
// double-buffered LDS staging pre-swizzled into WMMA fragment layout, with a
// software pipeline: global prefetch(k+1) -> wmma(k) -> convert/store(k+1),
// one workgroup barrier per K-step.

typedef __bf16        v16bf __attribute__((ext_vector_type(16)));
typedef float         v8f   __attribute__((ext_vector_type(8)));
typedef float         v4f   __attribute__((ext_vector_type(4)));
typedef unsigned int  v4u   __attribute__((ext_vector_type(4)));

#define NUM_E   64
#define MDIM    2048
#define NDIM    512
#define KDIM    512
#define MBLK    128
#define NBLK    128
#define KBLK    32
#define KSTEPS  (KDIM / KBLK)   // 16

// Split an fp32 pair into packed-bf16 hi and lo words (truncation split:
// hi = top16(x), lo = top16(x - hi)). ~17 effective mantissa bits combined.
__device__ __forceinline__ void pack2(float f0, float f1,
                                      unsigned* ph, unsigned* pl) {
  unsigned u0 = __float_as_uint(f0);
  unsigned u1 = __float_as_uint(f1);
  *ph = (u1 & 0xFFFF0000u) | (u0 >> 16);
  float r0 = f0 - __uint_as_float(u0 & 0xFFFF0000u);
  float r1 = f1 - __uint_as_float(u1 & 0xFFFF0000u);
  *pl = (__float_as_uint(r1) & 0xFFFF0000u) | (__float_as_uint(r0) >> 16);
}

union FragU {
  v16bf bf;
  v4u   c[2];
};

__global__ __launch_bounds__(256)
void moe_gemm_bf16x3(const float* __restrict__ X,
                     const float* __restrict__ W,
                     float* __restrict__ O) {
  // Two 32KB buffers, each in 16-byte chunks:
  // [A_hi 512][A_lo 512][B_hi 512][B_lo 512]  -> 64 KB total
  __shared__ v4u lds[4096];

  const int nb  = NDIM / NBLK;              // 4
  const int mb  = MDIM / MBLK;              // 16
  const int bid = blockIdx.x;
  const int e   = bid / (mb * nb);
  const int rem = bid % (mb * nb);
  const int m0  = (rem / nb) * MBLK;
  const int n0  = (rem % nb) * NBLK;

  const float* Xb = X + (size_t)(e * MDIM + m0) * KDIM;
  const float* Wb = W + (size_t)e * NDIM * KDIM + (size_t)n0 * KDIM;
  float*       Ob = O + (size_t)(e * MDIM + m0) * NDIM + n0;

  const int tid  = threadIdx.x;
  const int lane = tid & 31;
  const int w    = tid >> 5;   // wave 0..7
  const int wm   = w >> 2;     // 0..1  -> 64 rows of M
  const int wn   = w & 3;      // 0..3  -> 32 cols of N

  // ---- Loop-invariant staging descriptors (per thread, 2 chunks each A/B)
  const float* gA[2];
  const float* gB[2];
  int idxA[2], idxB[2];
#pragma unroll
  for (int s = 0; s < 2; ++s) {
    const int chunk = tid + s * 256;       // 0..511
    const int row   = chunk >> 2;          // 0..127
    const int j     = chunk & 3;           // K-subchunk of 8 floats
    gA[s] = Xb + (size_t)row * KDIM + j * 8;
    gB[s] = Wb + (size_t)row * KDIM + j * 8;
    // A 16x32 bf16 fragment layout: pos = j*16 + (row%16) within 64-chunk blk
    idxA[s] = (row >> 4) * 64 + j * 16 + (row & 15);
    // B 32x16 bf16 fragment layout: pos = (row%16)*4 + j within 64-chunk blk
    idxB[s] = 1024 + (row >> 4) * 64 + (row & 15) * 4 + j;
  }

  // ---- Loop-invariant fragment load indices
  int fA[4];
#pragma unroll
  for (int i = 0; i < 4; ++i) fA[i] = (wm * 4 + i) * 64 + lane;
  int fB[2];
#pragma unroll
  for (int j = 0; j < 2; ++j)
    fB[j] = 1024 + (wn * 2 + j) * 64 + (lane & 15) * 4 + (lane >> 4);

  v4f rA[2][2], rB[2][2];   // global prefetch registers (8 fp32 per chunk)

  auto load_tile = [&](int k0) {
#pragma unroll
    for (int s = 0; s < 2; ++s) {
      rA[s][0] = *(const v4f*)(gA[s] + k0);
      rA[s][1] = *(const v4f*)(gA[s] + k0 + 4);
      rB[s][0] = *(const v4f*)(gB[s] + k0);
      rB[s][1] = *(const v4f*)(gB[s] + k0 + 4);
    }
  };

  auto store_tile = [&](int buf) {
    const int boff = buf * 2048;
#pragma unroll
    for (int s = 0; s < 2; ++s) {
      unsigned ph[4], pl[4];
      pack2(rA[s][0][0], rA[s][0][1], &ph[0], &pl[0]);
      pack2(rA[s][0][2], rA[s][0][3], &ph[1], &pl[1]);
      pack2(rA[s][1][0], rA[s][1][1], &ph[2], &pl[2]);
      pack2(rA[s][1][2], rA[s][1][3], &ph[3], &pl[3]);
      v4u hiA = {ph[0], ph[1], ph[2], ph[3]};
      v4u loA = {pl[0], pl[1], pl[2], pl[3]};
      lds[boff + idxA[s]]       = hiA;
      lds[boff + 512 + idxA[s]] = loA;

      pack2(rB[s][0][0], rB[s][0][1], &ph[0], &pl[0]);
      pack2(rB[s][0][2], rB[s][0][3], &ph[1], &pl[1]);
      pack2(rB[s][1][0], rB[s][1][1], &ph[2], &pl[2]);
      pack2(rB[s][1][2], rB[s][1][3], &ph[3], &pl[3]);
      v4u hiB = {ph[0], ph[1], ph[2], ph[3]};
      v4u loB = {pl[0], pl[1], pl[2], pl[3]};
      lds[boff + idxB[s]]       = hiB;
      lds[boff + 512 + idxB[s]] = loB;
    }
  };

  v8f acc[4][2];
#pragma unroll
  for (int i = 0; i < 4; ++i)
#pragma unroll
    for (int j = 0; j < 2; ++j)
#pragma unroll
      for (int v = 0; v < 8; ++v) acc[i][j][v] = 0.0f;

  // ---- Pipeline prologue: stage K-step 0 into buffer 0
  load_tile(0);
  store_tile(0);
  __syncthreads();

  for (int kk = 0; kk < KSTEPS; ++kk) {
    const int cur = kk & 1;

    // Prefetch next K-step from HBM first (latency hides under WMMAs below)
    if (kk + 1 < KSTEPS) load_tile((kk + 1) * KBLK);

    // Fragment loads from current buffer (ds_load_b128 pairs)
    const int boff = cur * 2048;
    FragU ah[4], al[4], bh[2], bl[2];
#pragma unroll
    for (int i = 0; i < 4; ++i) {
      ah[i].c[0] = lds[boff + fA[i]];
      ah[i].c[1] = lds[boff + fA[i] + 32];
      al[i].c[0] = lds[boff + 512 + fA[i]];
      al[i].c[1] = lds[boff + 512 + fA[i] + 32];
    }
#pragma unroll
    for (int j = 0; j < 2; ++j) {
      bh[j].c[0] = lds[boff + fB[j]];
      bh[j].c[1] = lds[boff + fB[j] + 2];
      bl[j].c[0] = lds[boff + 512 + fB[j]];
      bl[j].c[1] = lds[boff + 512 + fB[j] + 2];
    }

    // 8 output tiles x 3 split-products = 24 v_wmma per K-step
#pragma unroll
    for (int i = 0; i < 4; ++i) {
#pragma unroll
      for (int j = 0; j < 2; ++j) {
        acc[i][j] = __builtin_amdgcn_wmma_f32_16x16x32_bf16(
            false, ah[i].bf, false, bh[j].bf, (short)0, acc[i][j], false, false);
        acc[i][j] = __builtin_amdgcn_wmma_f32_16x16x32_bf16(
            false, ah[i].bf, false, bl[j].bf, (short)0, acc[i][j], false, false);
        acc[i][j] = __builtin_amdgcn_wmma_f32_16x16x32_bf16(
            false, al[i].bf, false, bh[j].bf, (short)0, acc[i][j], false, false);
      }
    }

    // Convert + store next K-step into the other buffer (VALU co-executes
    // with the XDL WMMA pipe), then one barrier flips the buffers.
    if (kk + 1 < KSTEPS) store_tile(cur ^ 1);
    __syncthreads();
  }

  // ---- Epilogue: 16x16 f32 C/D layout: lane%16 = N col, VGPR v = M row
  //      (+8 rows for lanes 16..31)
#pragma unroll
  for (int i = 0; i < 4; ++i) {
#pragma unroll
    for (int j = 0; j < 2; ++j) {
      const int n = wn * 32 + j * 16 + (lane & 15);
#pragma unroll
      for (int v = 0; v < 8; ++v) {
        const int m = wm * 64 + i * 16 + v + 8 * (lane >> 4);
        Ob[(size_t)m * NDIM + n] = acc[i][j][v];
      }
    }
  }
}

extern "C" void kernel_launch(void* const* d_in, const int* in_sizes, int n_in,
                              void* d_out, int out_size, void* d_ws, size_t ws_size,
                              hipStream_t stream) {
  (void)in_sizes; (void)n_in; (void)d_ws; (void)ws_size; (void)out_size;
  const float* X = (const float*)d_in[0];   // [64*2048, 512]
  const float* W = (const float*)d_in[1];   // [64, 512, 512]
  float*       O = (float*)d_out;           // [64*2048, 512]

  const int grid = NUM_E * (MDIM / MBLK) * (NDIM / NBLK);  // 4096
  moe_gemm_bf16x3<<<grid, 256, 0, stream>>>(X, W, O);
}
